// LocalizedFiltering_9483287790026
// MI455X (gfx1250) — matrix operand order repaired and checked by reading, not compiled, not run on hardware
//
#include <hip/hip_runtime.h>

// ---------------------------------------------------------------------------
// LocalizedFiltering for MI455X (gfx1250, wave32, WMMA).
//
// Math: two GEMMs (each 137.5 GFLOP) dominate; compute-bound (AI ~450 F/B vs
// 23.3 TB/s HBM). fp32 inputs are decomposed x = hi(bf16) + lo(bf16) and each
// GEMM is computed as hi*Whi + hi*Wlo + lo*Whi with v_wmma_f32_16x16x32_bf16
// (3x bf16 cost ~= fp32 accuracy, ~2.7x faster than the 16x16x4 f32 WMMA path).
//
// Round 2: global->LDS staging now uses CDNA5 GLOBAL_LOAD_ASYNC_TO_LDS_B128
// (ASYNCcnt-tracked, no VGPR round trip) with double-buffered LDS so the next
// K-tile streams from HBM/L2 while the current tile's 24 WMMAs execute.
// ---------------------------------------------------------------------------

#define BS_   8
#define L_    2048
#define D_    2048
#define H_    1024
#define T_    (BS_ * L_)        // 16384 tokens
#define LP1   (L_ + 1)          // 2049 padded rows per sequence
#define MP    (BS_ * LP1)       // 16392 real padded rows
#define MPAD  16512             // 129 * 128 (grid-aligned row count)
#define N2    (2 * D_)          // 4096 (GEMM2 output cols before conv-combine)
#define EPS_  1e-6f

#define BM 128
#define BN 128
#define BK 32
#define LDK 40                  // BK + 8 pad: row stride 80B (16B aligned)

typedef __bf16 v16bf __attribute__((ext_vector_type(16)));
typedef float  v8f   __attribute__((ext_vector_type(8)));

union Frag16 { uint4 q[2]; v16bf v; };

// LDS aperture occupies addr[63:32]; addr[31:0] of a generic pointer to a
// __shared__ object is the wave-relative LDS byte offset (ISA 10.2), which is
// exactly what GLOBAL_LOAD_ASYNC_TO_LDS's VDST expects (an LDS byte offset).
__device__ __forceinline__ unsigned lds_off(const void* p) {
  return (unsigned)(unsigned long long)p;
}

__device__ __forceinline__ void async_b128(unsigned lds_dst, const void* gsrc) {
  asm volatile("global_load_async_to_lds_b128 %0, %1, off"
               :: "v"(lds_dst), "v"((unsigned long long)gsrc) : "memory");
}

// ---------------------------------------------------------------------------
// Prep: split fp32 weight [K,N] into bf16 hi/lo, stored TRANSPOSED [N][K] so
// B-fragment LDS reads are two contiguous uint4 per lane.
// ---------------------------------------------------------------------------
__global__ void lf_split_wT(const float* __restrict__ W, __bf16* __restrict__ Thi,
                            __bf16* __restrict__ Tlo, int K, int N) {
  long idx = (long)blockIdx.x * blockDim.x + threadIdx.x;
  long total = (long)K * N;
  if (idx >= total) return;
  int k = (int)(idx / N), n = (int)(idx % N);
  float w = W[idx];
  __bf16 h = (__bf16)w;
  Thi[(size_t)n * K + k] = h;
  Tlo[(size_t)n * K + k] = (__bf16)(w - (float)h);
}

// Padded A1 [MPAD, D]: row s*2049 = lf1_cache[pre_idx[s]], rows +1..+2048 = tokens,
// rows >= MP zero-filled. Split to bf16 hi/lo.
__global__ void lf_build_A1(const float* __restrict__ inp, const float* __restrict__ lf1,
                            const int* __restrict__ pre_idx,
                            __bf16* __restrict__ Ahi, __bf16* __restrict__ Alo) {
  long idx = (long)blockIdx.x * blockDim.x + threadIdx.x;
  if (idx >= (long)MPAD * D_) return;
  int r = (int)(idx / D_), k = (int)(idx % D_);
  float v = 0.0f;
  if (r < MP) {
    int s = r / LP1, j = r % LP1;
    v = (j == 0) ? lf1[(size_t)pre_idx[s] * D_ + k]
                 : inp[((size_t)s * L_ + (j - 1)) * D_ + k];
  }
  __bf16 h = (__bf16)v;
  Ahi[idx] = h;
  Alo[idx] = (__bf16)(v - (float)h);
}

// A2 fixed rows: cache rows (s*2049) from lf2, plus zero tail rows [MP, MPAD).
__global__ void lf_build_A2_fixed(const float* __restrict__ lf2, const int* __restrict__ pre_idx,
                                  __bf16* __restrict__ Ahi, __bf16* __restrict__ Alo) {
  int i = blockIdx.x * blockDim.x + threadIdx.x;
  const int cacheN = BS_ * H_;
  const int padN = (MPAD - MP) * H_;
  if (i >= cacheN + padN) return;
  int r, k; float v;
  if (i < cacheN) {
    int s = i / H_; k = i % H_;
    r = s * LP1;
    v = lf2[(size_t)pre_idx[s] * H_ + k];
  } else {
    int q = i - cacheN;
    r = MP + q / H_; k = q % H_;
    v = 0.0f;
  }
  size_t idx = (size_t)r * H_ + k;
  __bf16 h = (__bf16)v;
  Ahi[idx] = h;
  Alo[idx] = (__bf16)(v - (float)h);
}

// ---------------------------------------------------------------------------
// Tiled GEMM: C[MPAD,N] = (Ahi+Alo)[MPAD,K] @ (Bhi+Blo)[K,N], B given transposed
// [N][K]. 256 threads = 8 wave32; waves arranged 4(M) x 2(N); wave tile 32x64.
// Per K-chunk of 32: 2x4 fragment pairs x 3 WMMAs = 24 v_wmma_f32_16x16x32_bf16.
// Double-buffered LDS fed by GLOBAL_LOAD_ASYNC_TO_LDS_B128 (8 per thread per
// stage); s_wait_asynccnt 8 retires the previous stage (async loads complete
// in order) while the next stage's traffic is in flight.
// ---------------------------------------------------------------------------
__global__ void __launch_bounds__(256)
lf_gemm_bf16x3(const __bf16* __restrict__ Ahi, const __bf16* __restrict__ Alo,
               const __bf16* __restrict__ BThi, const __bf16* __restrict__ BTlo,
               float* __restrict__ C, int N, int K) {
  __shared__ __align__(16) __bf16 sAhi[2][BM][LDK];
  __shared__ __align__(16) __bf16 sAlo[2][BM][LDK];
  __shared__ __align__(16) __bf16 sBhi[2][BN][LDK];
  __shared__ __align__(16) __bf16 sBlo[2][BN][LDK];

  const int tid  = threadIdx.x;
  const int lane = tid & 31;
  const int wid  = tid >> 5;
  const int wm   = wid & 3;      // wave M position (0..3)
  const int wn   = wid >> 2;     // wave N position (0..1)
  const long m0  = (long)blockIdx.y * BM;
  const long n0  = (long)blockIdx.x * BN;

  v8f zero;
#pragma unroll
  for (int i = 0; i < 8; ++i) zero[i] = 0.0f;
  v8f acc[2][4];
#pragma unroll
  for (int mi = 0; mi < 2; ++mi)
#pragma unroll
    for (int nj = 0; nj < 4; ++nj) acc[mi][nj] = zero;

  const int srow = tid >> 1;            // staging row 0..127
  const int scol = (tid & 1) * 16;      // staging col chunk (16 bf16 = 32B)
  const size_t gabase = (size_t)(m0 + srow) * K + scol;   // A row base (elements)
  const size_t gbbase = (size_t)(n0 + srow) * K + scol;   // B row base (elements)
  const int mlane = lane & 15;
  const int hilane = lane >> 4;
  const int kbA = hilane * 8;           // A frag K base (elements)
  const int kbB = hilane * 8;           // B frag K base (elements)

  // Issue one stage's 8 async global->LDS b128 transfers for this thread.
  auto issue = [&](int buf, int k0) {
    async_b128(lds_off(&sAhi[buf][srow][scol]),     Ahi  + gabase + k0);
    async_b128(lds_off(&sAhi[buf][srow][scol + 8]), Ahi  + gabase + k0 + 8);
    async_b128(lds_off(&sAlo[buf][srow][scol]),     Alo  + gabase + k0);
    async_b128(lds_off(&sAlo[buf][srow][scol + 8]), Alo  + gabase + k0 + 8);
    async_b128(lds_off(&sBhi[buf][srow][scol]),     BThi + gbbase + k0);
    async_b128(lds_off(&sBhi[buf][srow][scol + 8]), BThi + gbbase + k0 + 8);
    async_b128(lds_off(&sBlo[buf][srow][scol]),     BTlo + gbbase + k0);
    async_b128(lds_off(&sBlo[buf][srow][scol + 8]), BTlo + gbbase + k0 + 8);
  };

  const int niter = K / BK;
  issue(0, 0);
  for (int it = 0; it < niter; ++it) {
    const int buf = it & 1;
    const int nxt = it + 1;
    if (nxt < niter) {
      issue(nxt & 1, nxt * BK);   // buffer last read in stage it-1; barrier at
                                  // end of that stage makes the overwrite safe
      asm volatile("s_wait_asynccnt 0x8" ::: "memory");  // stage-it loads done
    } else {
      asm volatile("s_wait_asynccnt 0x0" ::: "memory");
    }
    __syncthreads();              // stage-it LDS data visible to all waves

    Frag16 fa_h[2], fa_l[2], fb_h[4], fb_l[4];
#pragma unroll
    for (int mi = 0; mi < 2; ++mi) {
      const int m = wm * 32 + mi * 16 + mlane;
      fa_h[mi].q[0] = *(const uint4*)&sAhi[buf][m][kbA];
      fa_h[mi].q[1] = *(const uint4*)&sAhi[buf][m][kbA + 16];
      fa_l[mi].q[0] = *(const uint4*)&sAlo[buf][m][kbA];
      fa_l[mi].q[1] = *(const uint4*)&sAlo[buf][m][kbA + 16];
    }
#pragma unroll
    for (int nj = 0; nj < 4; ++nj) {
      const int n = wn * 64 + nj * 16 + mlane;
      fb_h[nj].q[0] = *(const uint4*)&sBhi[buf][n][kbB];
      fb_h[nj].q[1] = *(const uint4*)&sBhi[buf][n][kbB + 16];
      fb_l[nj].q[0] = *(const uint4*)&sBlo[buf][n][kbB];
      fb_l[nj].q[1] = *(const uint4*)&sBlo[buf][n][kbB + 16];
    }
#pragma unroll
    for (int mi = 0; mi < 2; ++mi) {
#pragma unroll
      for (int nj = 0; nj < 4; ++nj) {
        acc[mi][nj] = __builtin_amdgcn_wmma_f32_16x16x32_bf16(
            false, fa_h[mi].v, false, fb_h[nj].v, (short)0, acc[mi][nj], false, false);
        acc[mi][nj] = __builtin_amdgcn_wmma_f32_16x16x32_bf16(
            false, fa_h[mi].v, false, fb_l[nj].v, (short)0, acc[mi][nj], false, false);
        acc[mi][nj] = __builtin_amdgcn_wmma_f32_16x16x32_bf16(
            false, fa_l[mi].v, false, fb_h[nj].v, (short)0, acc[mi][nj], false, false);
      }
    }
    __syncthreads();              // all reads of buf done before it is re-filled
  }

  // C/D layout: VGPR i -> M = i + 8*(lane/16); N = lane%16.
#pragma unroll
  for (int mi = 0; mi < 2; ++mi) {
#pragma unroll
    for (int nj = 0; nj < 4; ++nj) {
      const long col = n0 + wn * 64 + nj * 16 + mlane;
      const long rbase = m0 + wm * 32 + mi * 16 + 8 * hilane;
#pragma unroll
      for (int i = 0; i < 8; ++i) {
        C[(size_t)(rbase + i) * N + col] = acc[mi][nj][i];
      }
    }
  }
}

// ---------------------------------------------------------------------------
// Epilogue 1: o1(s,j)[n] = C1[r][n] + C1[r+1][H+n] + b1[n], r = s*2049+j.
// Re-split to bf16 hi/lo into padded A2 at row r+1.
// ---------------------------------------------------------------------------
__global__ void lf_epilogue1(const float* __restrict__ C, const float* __restrict__ b1,
                             __bf16* __restrict__ A2hi, __bf16* __restrict__ A2lo) {
  long idx = (long)blockIdx.x * blockDim.x + threadIdx.x;
  if (idx >= (long)T_ * H_) return;
  int t = (int)(idx / H_), n = (int)(idx % H_);
  int s = t / L_, j = t % L_;
  long rc = (long)s * LP1 + j;
  float y = C[(size_t)rc * D_ + n] + C[(size_t)(rc + 1) * D_ + H_ + n] + b1[n];
  size_t d = (size_t)(rc + 1) * H_ + n;
  __bf16 h = (__bf16)y;
  A2hi[d] = h;
  A2lo[d] = (__bf16)(y - (float)h);
}

// Epilogue 2: o3 = conv-combine(C2) + b2 + residual(inputs) -> d_out (fp32).
__global__ void lf_epilogue2(const float* __restrict__ C, const float* __restrict__ b2,
                             const float* __restrict__ inp, float* __restrict__ out) {
  long idx = (long)blockIdx.x * blockDim.x + threadIdx.x;
  if (idx >= (long)T_ * D_) return;
  int t = (int)(idx / D_), n = (int)(idx % D_);
  int s = t / L_, j = t % L_;
  long rc = (long)s * LP1 + j;
  float y = C[(size_t)rc * N2 + n] + C[(size_t)(rc + 1) * N2 + D_ + n] + b2[n] + inp[idx];
  out[idx] = y;
}

// RMSNorm in place: one block per token row.
__global__ void lf_rmsnorm(float* __restrict__ out, const float* __restrict__ lnw) {
  __shared__ float red[256];
  const int t = blockIdx.x;
  const int tid = threadIdx.x;
  float ss = 0.0f;
  for (int n = tid; n < D_; n += 256) {
    float v = out[(size_t)t * D_ + n];
    ss += v * v;
  }
  red[tid] = ss;
  __syncthreads();
  for (int o = 128; o > 0; o >>= 1) {
    if (tid < o) red[tid] += red[tid + o];
    __syncthreads();
  }
  const float scale = rsqrtf(red[0] / (float)D_ + EPS_);
  for (int n = tid; n < D_; n += 256) {
    size_t i = (size_t)t * D_ + n;
    out[i] = out[i] * scale * lnw[n];
  }
}

// ---------------------------------------------------------------------------
extern "C" void kernel_launch(void* const* d_in, const int* in_sizes, int n_in,
                              void* d_out, int out_size, void* d_ws, size_t ws_size,
                              hipStream_t stream) {
  const float* inputs  = (const float*)d_in[0];
  const int*   pre_idx = (const int*)d_in[1];
  // d_in[2..6]: out_lf_indexs / input_lf_loc / out_lf_loc / inputs_loc /
  // outputs_loc — layout is closed-form (s*(L+1)+j), computed in-kernel.
  const float* lf1 = (const float*)d_in[7];
  const float* lf2 = (const float*)d_in[8];
  const float* W1  = (const float*)d_in[9];
  const float* W2  = (const float*)d_in[10];
  const float* b1  = (const float*)d_in[11];
  const float* b2  = (const float*)d_in[12];
  const float* lnw = (const float*)d_in[13];
  float* out = (float*)d_out;

  // Workspace carve (aligned 256B).
  char* ws = (char*)d_ws;
  size_t off = 0;
  auto carve = [&](size_t bytes) -> char* {
    char* p = ws + off;
    off += (bytes + 255) & ~(size_t)255;
    return p;
  };
  __bf16* A1hi  = (__bf16*)carve((size_t)MPAD * D_ * 2);
  __bf16* A1lo  = (__bf16*)carve((size_t)MPAD * D_ * 2);
  __bf16* W1Thi = (__bf16*)carve((size_t)D_ * D_ * 2);
  __bf16* W1Tlo = (__bf16*)carve((size_t)D_ * D_ * 2);
  __bf16* W2Thi = (__bf16*)carve((size_t)N2 * H_ * 2);
  __bf16* W2Tlo = (__bf16*)carve((size_t)N2 * H_ * 2);
  __bf16* A2hi  = (__bf16*)carve((size_t)MPAD * H_ * 2);
  __bf16* A2lo  = (__bf16*)carve((size_t)MPAD * H_ * 2);
  float*  Cbuf  = (float*)carve((size_t)MPAD * N2 * 4);
  (void)ws_size; (void)in_sizes; (void)n_in; (void)out_size;

  const int TB = 256;
  // Weight split + transpose (one-time per call, ~34 MB total traffic).
  {
    long tot = (long)D_ * D_;
    lf_split_wT<<<dim3((tot + TB - 1) / TB), dim3(TB), 0, stream>>>(W1, W1Thi, W1Tlo, D_, D_);
  }
  {
    long tot = (long)H_ * N2;
    lf_split_wT<<<dim3((tot + TB - 1) / TB), dim3(TB), 0, stream>>>(W2, W2Thi, W2Tlo, H_, N2);
  }
  // Padded A1 (scatter + bf16 hi/lo split).
  {
    long tot = (long)MPAD * D_;
    lf_build_A1<<<dim3((tot + TB - 1) / TB), dim3(TB), 0, stream>>>(inputs, lf1, pre_idx, A1hi, A1lo);
  }
  // A2 cache rows + zero tail.
  {
    long tot = (long)(BS_ + (MPAD - MP)) * H_;
    lf_build_A2_fixed<<<dim3((tot + TB - 1) / TB), dim3(TB), 0, stream>>>(lf2, pre_idx, A2hi, A2lo);
  }
  // GEMM1: [MPAD,2048] x [2048,2048] -> Cbuf (ld 2048).
  lf_gemm_bf16x3<<<dim3(D_ / BN, MPAD / BM), dim3(TB), 0, stream>>>(A1hi, A1lo, W1Thi, W1Tlo, Cbuf, D_, D_);
  // Conv shift-add + bias1 -> padded A2 (bf16 hi/lo).
  {
    long tot = (long)T_ * H_;
    lf_epilogue1<<<dim3((tot + TB - 1) / TB), dim3(TB), 0, stream>>>(Cbuf, b1, A2hi, A2lo);
  }
  // GEMM2: [MPAD,1024] x [1024,4096] -> Cbuf (ld 4096).
  lf_gemm_bf16x3<<<dim3(N2 / BN, MPAD / BM), dim3(TB), 0, stream>>>(A2hi, A2lo, W2Thi, W2Tlo, Cbuf, N2, H_);
  // Conv shift-add + bias2 + residual -> d_out.
  {
    long tot = (long)T_ * D_;
    lf_epilogue2<<<dim3((tot + TB - 1) / TB), dim3(TB), 0, stream>>>(Cbuf, b2, inputs, out);
  }
  // RMSNorm in place.
  lf_rmsnorm<<<dim3(T_), dim3(TB), 0, stream>>>(out, lnw);
}
